// LTMModule_60885456388383
// MI455X (gfx1250) — matrix operand order repaired and checked by reading, not compiled, not run on hardware
//
#include <hip/hip_runtime.h>

typedef __attribute__((ext_vector_type(16))) __bf16 v16bf;
typedef __attribute__((ext_vector_type(8)))  float  v8f;

#define N_SLOTS 8192
#define KEY_DIM 64
#define VAL_DIM 64
#define M_TOTAL (8 * 2048)      // B*S = 16384 queries
#define TOPK 8
#define WAVES_PER_BLOCK 8
#define BLOCK (WAVES_PER_BLOCK * 32)
#define QTILE 16

// Order-preserving pack: float -> monotone u32, low 13 bits carry the slot id.
static __device__ __forceinline__ uint32_t pack_key(float v, uint32_t slot) {
  uint32_t b = __float_as_uint(v);
  uint32_t m = b ^ ((uint32_t)((int32_t)b >> 31) | 0x80000000u);
  return (m & 0xFFFFE000u) | slot;
}

// Sorted-descending top-8 insert: 1-op screen + 16-op max/min bubble network.
static __device__ __forceinline__ void insert8(uint32_t (&t)[8], uint32_t key) {
  if (key <= t[7]) return;
  uint32_t x = key;
#pragma unroll
  for (int k = 0; k < 8; ++k) {
    uint32_t hi = t[k] >= x ? t[k] : x;
    uint32_t lo = t[k] >= x ? x : t[k];
    t[k] = hi; x = lo;
  }
}

template<int BASE>
static __device__ __forceinline__ void put8(v16bf &d, float4 a, float4 b) {
  d[BASE+0]=(__bf16)a.x; d[BASE+1]=(__bf16)a.y; d[BASE+2]=(__bf16)a.z; d[BASE+3]=(__bf16)a.w;
  d[BASE+4]=(__bf16)b.x; d[BASE+5]=(__bf16)b.y; d[BASE+6]=(__bf16)b.z; d[BASE+7]=(__bf16)b.w;
}

__global__ __launch_bounds__(BLOCK)
void ltm_topk_attend(const float* __restrict__ queries,
                     const float* __restrict__ keys,
                     const float* __restrict__ vals,
                     const float* __restrict__ fast,
                     const float* __restrict__ ts,
                     float* __restrict__ outWV,
                     float* __restrict__ outIdx,
                     float* __restrict__ outTs) {
  __shared__ uint32_t sm[BLOCK * TOPK];

  const int tid  = threadIdx.x;
  const int wave = tid >> 5;
  const int lane = tid & 31;
  const int row  = lane & 15;
  const int half = lane >> 4;

  const int m0 = (blockIdx.x * WAVES_PER_BLOCK + wave) * QTILE;
  const int m  = m0 + row;   // this lane's query column (B-matrix N index)

  // B fragments (queries, resident in regs): B(32x16) bf16, lane=col N, K = 8*half + e (e<8), 16 + 8*half + (e-8) (e>=8).
  const float4* q4 = (const float4*)(queries + (size_t)m * KEY_DIM);
  v16bf bq0, bq1;
  {
    const int f = 2 * half;
    put8<0>(bq0, q4[f+0],   q4[f+1]);   put8<8>(bq0, q4[f+4],   q4[f+5]);
    put8<0>(bq1, q4[8+f+0], q4[8+f+1]); put8<8>(bq1, q4[8+f+4], q4[8+f+5]);
  }

  uint32_t top[8] = {0,0,0,0,0,0,0,0};

  // Phase 1: sim^T tile = keys(16x64) x Q^T(64x16) via 2 chained bf16 WMMAs,
  // fused per-lane top-8 over this lane's 8 slots per tile.
  const int f0 = 2 * half;  // A-fragment float4 base: K runs {8h, 8h+16, 8h+32, 8h+48}
  for (int n0 = 0; n0 < N_SLOTS; n0 += 16) {
    const float4* k4 = (const float4*)(keys + (size_t)(n0 + row) * KEY_DIM);
    if (n0 + 16 < N_SLOTS)
      __builtin_prefetch(keys + (size_t)(n0 + 16 + row) * KEY_DIM, 0, 0);

    v16bf a0, a1;
    put8<0>(a0, k4[f0+0],  k4[f0+1]);   put8<8>(a0, k4[f0+4],  k4[f0+5]);
    put8<0>(a1, k4[f0+8],  k4[f0+9]);   put8<8>(a1, k4[f0+12], k4[f0+13]);

    v8f acc = {0.f,0.f,0.f,0.f,0.f,0.f,0.f,0.f};
    acc = __builtin_amdgcn_wmma_f32_16x16x32_bf16(false, a0, false, bq0, (short)0, acc, false, false);
    acc = __builtin_amdgcn_wmma_f32_16x16x32_bf16(false, a1, false, bq1, (short)0, acc, false, false);

    // Tile-level screen: pack(max, all-ones idx) is a strict upper bound on any
    // key from this tile, so a failing compare rejects all 8 values at once.
    float mxv = acc[0];
#pragma unroll
    for (int r = 1; r < 8; ++r) mxv = fmaxf(mxv, acc[r]);
    if (pack_key(mxv, 0x1FFFu) > top[7]) {
#pragma unroll
      for (int r = 0; r < 8; ++r) {
        const uint32_t slot = (uint32_t)(n0 + 8 * half + r);   // C-layout: M = r + 8*half
        insert8(top, pack_key(acc[r], slot));
      }
    }
  }

  // Merge lane L / L+16 half-space lists via LDS; lane<16 owns query m0+L's final list.
#pragma unroll
  for (int i = 0; i < 8; ++i) sm[tid * 8 + i] = top[i];
  __syncthreads();
  if (half == 0) {
#pragma unroll
    for (int i = 0; i < 8; ++i) insert8(top, sm[(tid + 16) * 8 + i]);
  }
  __syncthreads();
  if (half == 0) {
#pragma unroll
    for (int i = 0; i < 8; ++i) sm[tid * 8 + i] = top[i];
  }
  __syncthreads();
  uint32_t idxs[8];
#pragma unroll
  for (int i = 0; i < 8; ++i) idxs[i] = sm[(tid & ~16) * 8 + i] & 0x1FFFu;

  // Phase 2: exact f32 rescore (both half-lanes split the 64-dim in two),
  // softmax over 8, weighted gather of (vals + fast).
  const int dbase = 32 * half;
  float4 qc[8];
  {
    const float4* qp = (const float4*)(queries + (size_t)m * KEY_DIM + dbase);
#pragma unroll
    for (int t = 0; t < 8; ++t) qc[t] = qp[t];
  }

  float sims[8];
#pragma unroll
  for (int j = 0; j < 8; ++j) {
    const float4* kp = (const float4*)(keys + (size_t)idxs[j] * KEY_DIM + dbase);
    float p = 0.f;
#pragma unroll
    for (int t = 0; t < 8; ++t) {
      float4 a = qc[t], b = kp[t];
      p += a.x*b.x + a.y*b.y + a.z*b.z + a.w*b.w;
    }
    // combine the two 32-dim halves: ds_swizzle SWAPX16 (xor 0x10)
    float o = __int_as_float(__builtin_amdgcn_ds_swizzle(__float_as_int(p), 0x401F));
    sims[j] = (p + o) * 0.125f;   // KEY_DIM^-0.5
  }
  float mx = sims[0];
#pragma unroll
  for (int j = 1; j < 8; ++j) mx = fmaxf(mx, sims[j]);
  float e[8], sum = 0.f;
#pragma unroll
  for (int j = 0; j < 8; ++j) { e[j] = __expf(sims[j] - mx); sum += e[j]; }
  const float inv = 1.f / sum;

#pragma unroll
  for (int j = 0; j < 8; ++j) {
    const float w = e[j] * inv;
    const uint32_t id = idxs[j];
    const float4* vv = (const float4*)(vals + (size_t)id * VAL_DIM + dbase);
    const float4* fv = (const float4*)(fast + (size_t)id * VAL_DIM + dbase);
    float4* ov = (float4*)(outWV + ((size_t)m * TOPK + j) * VAL_DIM + dbase);
#pragma unroll
    for (int t = 0; t < 8; ++t) {
      float4 a = vv[t], b = fv[t];
      ov[t] = make_float4((a.x+b.x)*w, (a.y+b.y)*w, (a.z+b.z)*w, (a.w+b.w)*w);
    }
    if (half == 0) outIdx[(size_t)m * TOPK + j] = (float)id;
    else           outTs [(size_t)m * TOPK + j] = ts[id];
  }
}

extern "C" void kernel_launch(void* const* d_in, const int* in_sizes, int n_in,
                              void* d_out, int out_size, void* d_ws, size_t ws_size,
                              hipStream_t stream) {
  (void)in_sizes; (void)n_in; (void)out_size; (void)d_ws; (void)ws_size;
  const float* queries = (const float*)d_in[0];
  const float* keys    = (const float*)d_in[1];
  const float* vals    = (const float*)d_in[2];
  const float* fast    = (const float*)d_in[3];
  const float* ts      = (const float*)d_in[4];

  float* out    = (float*)d_out;
  float* outWV  = out;                                           // [16384, 8, 64]
  float* outIdx = out + (size_t)M_TOTAL * TOPK * VAL_DIM;        // [16384, 8]
  float* outTs  = outIdx + (size_t)M_TOTAL * TOPK;               // [16384, 8]

  const int blocks = M_TOTAL / (QTILE * WAVES_PER_BLOCK);        // 128 blocks x 256 thr
  ltm_topk_attend<<<blocks, BLOCK, 0, stream>>>(queries, keys, vals, fast, ts,
                                                outWV, outIdx, outTs);
}
